// OperatorAttention_41120016892133
// MI455X (gfx1250) — compile-verified, matrix operand order checked by reading, not run on hardware
//
#include <hip/hip_runtime.h>
#include <cstdint>

// Problem constants (match reference)
#define B_    2
#define T_    8192
#define DM_   2048
#define H_    16
#define HKV_  8
#define DH_   128
#define ML_   64
#define SCALE_ 0.08838834764831845f   // 1/sqrt(128)

typedef __attribute__((ext_vector_type(16))) __bf16 v16bf;
typedef __attribute__((ext_vector_type(8)))  float  v8f;

__device__ __forceinline__ __bf16 f2bf(float f) {
  union { float f; unsigned u; } a; a.f = f;
  unsigned u = a.u + 0x7FFFu + ((a.u >> 16) & 1u);   // RNE
  union { unsigned short s; __bf16 b; } o; o.s = (unsigned short)(u >> 16);
  return o.b;
}

__device__ __forceinline__ v8f v8f_zero() {
  v8f z;
#pragma unroll
  for (int i = 0; i < 8; ++i) z[i] = 0.f;
  return z;
}

// ---- CDNA5 async global->LDS DMA (tracked by ASYNCcnt, no VGPR round-trip).
// Low 32 bits of a generic pointer into __shared__ = LDS byte offset (aperture).
__device__ __forceinline__ void async_copy_b128(void* lds_dst, const void* gsrc) {
  unsigned loff = (unsigned)(uintptr_t)lds_dst;
  asm volatile("global_load_async_to_lds_b128 %0, %1, off"
               :: "v"(loff), "v"(gsrc) : "memory");
}
__device__ __forceinline__ void wait_async0() {
  asm volatile("s_wait_asynccnt 0x0" ::: "memory");
}

// 16x32 bf16 fragment loader (A and B share the same per-lane packing).
// Lane L: row = row0 + (L&15); VGPR0-3 hold K = k0 + 8*(L>>4) + 0..7,
// VGPR4-7 hold K = k0 + 16 + 8*(L>>4) + 0..7. Requires k-contiguous storage.
__device__ __forceinline__ v16bf load_frag16(const __bf16* base, int ld,
                                             int row0, int k0, int lane) {
  int r  = row0 + (lane & 15);
  int h8 = (lane >> 4) << 3;
  const __bf16* p = base + (size_t)r * ld + k0 + h8;
  v16bf x;
#pragma unroll
  for (int i = 0; i < 8; ++i) { x[i] = p[i]; x[8 + i] = p[16 + i]; }
  return x;
}

// 8 WMMAs over one 64x64x64 LDS tile (per-wave 16-row strip of C).
__device__ __forceinline__ void mma_tile(const __bf16* bufA, const __bf16* bufB,
                                         int wave, int lane, v8f acc[4]) {
#pragma unroll
  for (int ks = 0; ks < 2; ++ks) {
    v16bf a = load_frag16(bufA, 64, wave * 16, ks * 32, lane);
#pragma unroll
    for (int j = 0; j < 4; ++j) {
      v16bf b = load_frag16(bufB, 64, j * 16, ks * 32, lane);
      acc[j] = __builtin_amdgcn_wmma_f32_16x16x32_bf16(
          false, a, false, b, (short)0, acc[j], false, false);
    }
  }
}

// Per-thread async DMA of one 64x64 A-tile + B-tile slice (64B each).
__device__ __forceinline__ void issue_tile(__bf16* dA, __bf16* dB,
                                           const __bf16* ga, const __bf16* gb) {
#pragma unroll
  for (int i = 0; i < 4; ++i) {
    async_copy_b128(dA + i * 8, ga + i * 8);
    async_copy_b128(dB + i * 8, gb + i * 8);
  }
}

// ---------------------------------------------------------------- cast f32->bf16
__global__ void cast_f32_bf16(const float* __restrict__ in,
                              __bf16* __restrict__ out, size_t n) {
  size_t i = (size_t)blockIdx.x * blockDim.x + threadIdx.x;
  size_t stride = (size_t)gridDim.x * blockDim.x;
  for (; i < n; i += stride) out[i] = f2bf(in[i]);
}

// ------------------------------------- cast + transpose: in[K][N] -> out[N][K]
__global__ __launch_bounds__(256) void cast_transpose_bf16(
    const float* __restrict__ in, __bf16* __restrict__ out, int K, int N) {
  __shared__ float tile[32][33];
  const int tx = threadIdx.x, ty = threadIdx.y;          // (32,8)
  const int k0 = blockIdx.x * 32, n0 = blockIdx.y * 32;
  for (int i = ty; i < 32; i += 8)
    tile[i][tx] = in[(size_t)(k0 + i) * N + n0 + tx];
  __syncthreads();
  for (int i = ty; i < 32; i += 8)
    out[(size_t)(n0 + i) * K + k0 + tx] = f2bf(tile[tx][i]);
}

// ---------------------------- GEMM: C(f32) = A(bf16, MxK) · BT(bf16, NxK)ᵀ
// BT is N-major/k-contiguous. Block 128 threads = 4 waves; 64x64 C tile, BK=64.
// Double-buffered async DMA pipeline, unrolled 2 tiles/iter (K % 128 == 0).
__global__ __launch_bounds__(128) void gemm_bf16_f32(
    const __bf16* __restrict__ A, const __bf16* __restrict__ BT,
    float* __restrict__ C, int M, int N, int K) {
  __shared__ __bf16 sA[2][64 * 64];    // [m][k]  8KB each
  __shared__ __bf16 sB[2][64 * 64];    // [n][k]  8KB each
  const int tid = threadIdx.x, lane = tid & 31, wave = tid >> 5;
  const int m0 = blockIdx.y * 64, n0 = blockIdx.x * 64;
  v8f acc[4];
#pragma unroll
  for (int j = 0; j < 4; ++j) acc[j] = v8f_zero();

  const int rl = tid >> 1, cl = (tid & 1) * 32;   // 32 elems (64B) per thread/tile
  const int lo = rl * 64 + cl;                    // this thread's LDS slot
  const __bf16* ga0 = A  + (size_t)(m0 + rl) * K + cl;
  const __bf16* gb0 = BT + (size_t)(n0 + rl) * K + cl;

  // prime buffer 0 with tile kk=0
  issue_tile(&sA[0][lo], &sB[0][lo], ga0, gb0);
  wait_async0();
  __syncthreads();

  for (int kk = 0; kk < K; kk += 128) {
    // phase 0: DMA tile kk+64 into buf1 while computing buf0
    issue_tile(&sA[1][lo], &sB[1][lo], ga0 + kk + 64, gb0 + kk + 64);
    mma_tile(sA[0], sB[0], wave, lane, acc);
    wait_async0();
    __syncthreads();
    // phase 1: DMA tile kk+128 into buf0 (if any) while computing buf1
    if (kk + 128 < K)
      issue_tile(&sA[0][lo], &sB[0][lo], ga0 + kk + 128, gb0 + kk + 128);
    mma_tile(sA[1], sB[1], wave, lane, acc);
    wait_async0();
    __syncthreads();
  }

  const int ln = lane & 15, lh = lane >> 4;
#pragma unroll
  for (int j = 0; j < 4; ++j) {
    int col = n0 + j * 16 + ln;
#pragma unroll
    for (int vr = 0; vr < 8; ++vr) {
      int row = m0 + wave * 16 + vr + 8 * lh;
      C[(size_t)row * N + col] = acc[j][vr];
    }
  }
}

// --------------------------------- landmarks + kernel_proj: z + silu(z·Wkp1)·Wkp2
__global__ __launch_bounds__(128) void landmark_proj(
    const float* __restrict__ src, int nheads,
    const float* __restrict__ Wkp1, const float* __restrict__ Wkp2,
    float* __restrict__ land, __bf16* __restrict__ landbf) {
  __shared__ float z[64 * 128];
  __shared__ float ss[64];
  const int h = blockIdx.x, b = blockIdx.y, tid = threadIdx.x;
  { int r = tid >> 1, c0 = (tid & 1) * 64;
    const float* g = src + ((size_t)(b * T_ + r * 128) * nheads + h) * 128 + c0;
    for (int i = 0; i < 64; ++i) z[r * 128 + c0 + i] = g[i];
  }
  __syncthreads();
  size_t obase = (size_t)(b * nheads + h) * 64 * 128;
  for (int m = 0; m < 64; ++m) {
    if (tid < 64) {
      float s = 0.f;
      for (int d = 0; d < 128; ++d) s += z[m * 128 + d] * Wkp1[d * 64 + tid];
      ss[tid] = s / (1.f + expf(-s));        // silu
    }
    __syncthreads();
    float o = z[m * 128 + tid];
    for (int r = 0; r < 64; ++r) o += ss[r] * Wkp2[r * 128 + tid];
    land[obase + m * 128 + tid] = o;
    if (landbf) landbf[obase + m * 128 + tid] = f2bf(o);
    __syncthreads();
  }
}

// ------------------- K_mm = softmax(q_land·k_landᵀ·s); Z = iterative pinv(K_mm)
__global__ __launch_bounds__(256) void kmm_pinv(
    const float* __restrict__ qland, const float* __restrict__ kland,
    float* __restrict__ zinv) {
  __shared__ float A[64 * 64], Z[64 * 64], T1[64 * 64];
  __shared__ float red1[64], red2[64];
  __shared__ float vsh;
  const int h = blockIdx.x, b = blockIdx.y, tid = threadIdx.x;
  const float* ql = qland + (size_t)(b * H_ + h) * 64 * 128;
  const float* kl = kland + (size_t)(b * HKV_ + (h >> 1)) * 64 * 128;
  for (int idx = tid; idx < 4096; idx += 256) {
    int m = idx >> 6, n = idx & 63;
    float s = 0.f;
#pragma unroll 4
    for (int d = 0; d < 128; ++d) s += ql[m * 128 + d] * kl[n * 128 + d];
    A[idx] = s * SCALE_;
  }
  __syncthreads();
  if (tid < 64) {                              // row softmax
    float mx = -1e30f;
    for (int n = 0; n < 64; ++n) mx = fmaxf(mx, A[tid * 64 + n]);
    float sm = 0.f;
    for (int n = 0; n < 64; ++n) { float p = expf(A[tid * 64 + n] - mx); A[tid * 64 + n] = p; sm += p; }
    float inv = 1.f / sm;
    for (int n = 0; n < 64; ++n) A[tid * 64 + n] *= inv;
  }
  __syncthreads();
  if (tid < 64) {                              // 1-norm / inf-norm
    float cs = 0.f, rs = 0.f;
    for (int m = 0; m < 64; ++m) cs += fabsf(A[m * 64 + tid]);
    for (int n = 0; n < 64; ++n) rs += fabsf(A[tid * 64 + n]);
    red1[tid] = cs; red2[tid] = rs;
  }
  __syncthreads();
  if (tid == 0) {
    float n1 = 0.f, ni = 0.f;
    for (int i = 0; i < 64; ++i) { n1 = fmaxf(n1, red1[i]); ni = fmaxf(ni, red2[i]); }
    vsh = 1.f / fmaxf(n1 * ni, 1e-7f);
  }
  __syncthreads();
  for (int idx = tid; idx < 4096; idx += 256) {
    int m = idx >> 6, n = idx & 63;
    Z[idx] = A[n * 64 + m] * vsh;              // Z0 = Aᵀ·v
  }
  __syncthreads();
  for (int it = 0; it < 6; ++it) {             // Z = 2Z - Z·A·Z
    for (int idx = tid; idx < 4096; idx += 256) {
      int m = idx >> 6, n = idx & 63;
      float s = 0.f;
      for (int k = 0; k < 64; ++k) s += A[m * 64 + k] * Z[k * 64 + n];
      T1[idx] = s;
    }
    __syncthreads();
    float t2[16]; int c = 0;
    for (int idx = tid; idx < 4096; idx += 256, ++c) {
      int m = idx >> 6, n = idx & 63;
      float s = 0.f;
      for (int k = 0; k < 64; ++k) s += Z[m * 64 + k] * T1[k * 64 + n];
      t2[c] = s;
    }
    __syncthreads();
    c = 0;
    for (int idx = tid; idx < 4096; idx += 256, ++c) Z[idx] = 2.f * Z[idx] - t2[c];
    __syncthreads();
  }
  for (int idx = tid; idx < 4096; idx += 256)
    zinv[(size_t)(b * H_ + h) * 4096 + idx] = Z[idx];
}

// -------- KV = softmax_t(q_land·kᵀ·s) · v, flash-style online softmax over T.
__global__ __launch_bounds__(128) void kmk_softmax_v(
    const float* __restrict__ kf, const float* __restrict__ vf,
    const float* __restrict__ qland, float* __restrict__ kvout) {
  __shared__ float sq[128], red[128], wbuf[128];
  const int m = blockIdx.x, h = blockIdx.y, b = blockIdx.z, tid = threadIdx.x;
  const int kvh = h >> 1;
  sq[tid] = qland[((size_t)(b * H_ + h) * 64 + m) * 128 + tid];
  __syncthreads();
  float acc = 0.f, ssum = 0.f, mold = -1e30f;
  for (int c0 = 0; c0 < T_; c0 += 128) {
    const float* kp = kf + ((size_t)(b * T_ + c0 + tid) * HKV_ + kvh) * 128;
    float dot = 0.f;
#pragma unroll 4
    for (int d = 0; d < 128; ++d) dot += sq[d] * kp[d];
    float l = dot * SCALE_;
    red[tid] = l; __syncthreads();
    for (int s = 64; s > 0; s >>= 1) { if (tid < s) red[tid] = fmaxf(red[tid], red[tid + s]); __syncthreads(); }
    float mnew = fmaxf(mold, red[0]);
    __syncthreads();
    float p = expf(l - mnew);
    wbuf[tid] = p;
    red[tid] = p; __syncthreads();
    for (int s = 64; s > 0; s >>= 1) { if (tid < s) red[tid] += red[tid + s]; __syncthreads(); }
    float csum = red[0];
    float f = expf(mold - mnew);
    ssum = ssum * f + csum;
    acc *= f;
    const float* vbase = vf + ((size_t)(b * T_ + c0) * HKV_ + kvh) * 128 + tid;
    for (int t = 0; t < 128; ++t)              // coalesced across tid (=dim)
      acc += wbuf[t] * vbase[(size_t)t * HKV_ * 128];
    mold = mnew;
    __syncthreads();
  }
  kvout[((size_t)(b * H_ + h) * 64 + m) * 128 + tid] = acc / ssum;
}

// ------------------- W2ᵀ = (Zinv · KV)ᵀ, stored [d][m] bf16 (k-contiguous B op)
__global__ __launch_bounds__(256) void apply_pinv(
    const float* __restrict__ zinv, const float* __restrict__ kv,
    __bf16* __restrict__ w2tbf) {
  __shared__ float sZ[4096];
  const int h = blockIdx.x, b = blockIdx.y, tid = threadIdx.x;
  size_t zb = (size_t)(b * H_ + h) * 4096;
  size_t kb = (size_t)(b * H_ + h) * 64 * 128;
  for (int i = tid; i < 4096; i += 256) sZ[i] = zinv[zb + i];
  __syncthreads();
  for (int idx = tid; idx < 8192; idx += 256) {
    int m = idx >> 7, d = idx & 127;
    float s = 0.f;
    for (int r = 0; r < 64; ++r) s += sZ[m * 64 + r] * kv[kb + r * 128 + d];
    w2tbf[kb + (size_t)d * 64 + m] = f2bf(s);   // transposed store
  }
}

// ------- Fused: attn = softmax(Q·K_landᵀ·s) · W2, WMMA for both matmuls.
// Block 128 threads (4 waves). bf16 operand fills use async global->LDS DMA.
__global__ __launch_bounds__(128) void kqm_out(
    const float* __restrict__ qf, const __bf16* __restrict__ klandbf,
    const __bf16* __restrict__ w2tbf, __bf16* __restrict__ attnbf) {
  __shared__ __attribute__((aligned(16))) char smem[65536];
  __bf16* sQ   = (__bf16*)smem;                 // 64x128 bf16 (16KB)
  __bf16* sKL  = (__bf16*)(smem + 16384);       // 64x128 bf16 (16KB)
  __bf16* sW2T = (__bf16*)(smem + 32768);       // 128x64 bf16, [d][m] (16KB)
  float*  sS   = (float*)(smem + 49152);        // 64x64 f32 (16KB)
  __bf16* sP   = (__bf16*)smem;                 // overlays dead sQ (8KB)

  const int tb = blockIdx.x, h = blockIdx.y, b = blockIdx.z;
  const int tid = threadIdx.x, lane = tid & 31, wave = tid >> 5;
  const int t0 = tb * 64;
  // async DMA the two bf16 operands (128B per thread each)
  { int r = tid >> 1, c0 = (tid & 1) * 64;
    const __bf16* gk = klandbf + ((size_t)(b * HKV_ + (h >> 1)) * 64 + r) * 128 + c0;
    const __bf16* gw = w2tbf + ((size_t)(b * H_ + h) * 128 + tid) * 64;
#pragma unroll
    for (int i = 0; i < 8; ++i) {
      async_copy_b128(&sKL[r * 128 + c0 + i * 8], gk + i * 8);
      async_copy_b128(&sW2T[tid * 64 + i * 8], gw + i * 8);
    }
  }
  // Q: f32 -> bf16 conversion in flight
  { int r = tid >> 1, c0 = (tid & 1) * 64;
    const float* g = qf + (size_t)(b * T_ + t0 + r) * (H_ * DH_) + h * DH_ + c0;
    for (int i = 0; i < 64; ++i) sQ[r * 128 + c0 + i] = f2bf(g[i]);
  }
  wait_async0();
  __syncthreads();

  // S = Q(64x128) · KLᵀ -> 64x64, each wave a 16-row strip
  v8f accS[4];
#pragma unroll
  for (int j = 0; j < 4; ++j) accS[j] = v8f_zero();
#pragma unroll
  for (int ks = 0; ks < 4; ++ks) {
    v16bf a = load_frag16(sQ, 128, wave * 16, ks * 32, lane);
#pragma unroll
    for (int j = 0; j < 4; ++j) {
      v16bf bb = load_frag16(sKL, 128, j * 16, ks * 32, lane);
      accS[j] = __builtin_amdgcn_wmma_f32_16x16x32_bf16(
          false, a, false, bb, (short)0, accS[j], false, false);
    }
  }
  const int ln = lane & 15, lh = lane >> 4;
#pragma unroll
  for (int j = 0; j < 4; ++j)
#pragma unroll
    for (int vr = 0; vr < 8; ++vr)
      sS[(wave * 16 + vr + 8 * lh) * 64 + j * 16 + ln] = accS[j][vr];
  __syncthreads();

  if (tid < 64) {                               // row softmax (scale then softmax)
    float mx = -1e30f;
    for (int n = 0; n < 64; ++n) mx = fmaxf(mx, sS[tid * 64 + n]);
    float sm = 0.f;
    for (int n = 0; n < 64; ++n) {
      float p = expf((sS[tid * 64 + n] - mx) * SCALE_);
      sS[tid * 64 + n] = p; sm += p;
    }
    float inv = 1.f / sm;
    for (int n = 0; n < 64; ++n) sP[tid * 64 + n] = f2bf(sS[tid * 64 + n] * inv);
  }
  __syncthreads();

  // O = P(64x64) · W2(64x128)
  v8f accO[8];
#pragma unroll
  for (int j = 0; j < 8; ++j) accO[j] = v8f_zero();
#pragma unroll
  for (int ks = 0; ks < 2; ++ks) {
    v16bf a = load_frag16(sP, 64, wave * 16, ks * 32, lane);
#pragma unroll
    for (int j = 0; j < 8; ++j) {
      v16bf bb = load_frag16(sW2T, 64, j * 16, ks * 32, lane);
      accO[j] = __builtin_amdgcn_wmma_f32_16x16x32_bf16(
          false, a, false, bb, (short)0, accO[j], false, false);
    }
  }
#pragma unroll
  for (int j = 0; j < 8; ++j)
#pragma unroll
    for (int vr = 0; vr < 8; ++vr) {
      int mrow = wave * 16 + vr + 8 * lh;
      attnbf[(size_t)(b * T_ + t0 + mrow) * (H_ * DH_) + h * DH_ + j * 16 + ln] =
          f2bf(accO[j][vr]);
    }
}

// ================================================================ host launcher
extern "C" void kernel_launch(void* const* d_in, const int* in_sizes, int n_in,
                              void* d_out, int out_size, void* d_ws, size_t ws_size,
                              hipStream_t stream) {
  (void)in_sizes; (void)n_in; (void)out_size; (void)ws_size;
  const float* x    = (const float*)d_in[0];
  const float* Wq   = (const float*)d_in[1];
  const float* Wk   = (const float*)d_in[2];
  const float* Wv   = (const float*)d_in[3];
  const float* Wo   = (const float*)d_in[4];
  const float* Wkp1 = (const float*)d_in[5];
  const float* Wkp2 = (const float*)d_in[6];
  float* out = (float*)d_out;

  char* w = (char*)d_ws;
  size_t off = 0;
  auto carve = [&](size_t bytes) -> char* {
    char* p = w + off;
    off = (off + bytes + 255) & ~(size_t)255;
    return p;
  };
  const size_t NT = (size_t)B_ * T_;                    // 16384 tokens
  __bf16* xbf     = (__bf16*)carve(NT * 2048 * 2);
  __bf16* wqt     = (__bf16*)carve((size_t)2048 * 2048 * 2);   // [N][K]
  __bf16* wkt     = (__bf16*)carve((size_t)1024 * 2048 * 2);
  __bf16* wvt     = (__bf16*)carve((size_t)1024 * 2048 * 2);
  __bf16* wot     = (__bf16*)carve((size_t)2048 * 2048 * 2);
  float*  qf      = (float*)carve(NT * 2048 * 4);
  float*  kf      = (float*)carve(NT * 1024 * 4);
  float*  vf      = (float*)carve(NT * 1024 * 4);
  __bf16* attnbf  = (__bf16*)carve(NT * 2048 * 2);
  float*  qland   = (float*)carve((size_t)B_ * H_ * 64 * 128 * 4);
  float*  kland   = (float*)carve((size_t)B_ * HKV_ * 64 * 128 * 4);
  __bf16* klandbf = (__bf16*)carve((size_t)B_ * HKV_ * 64 * 128 * 2);
  float*  zinv    = (float*)carve((size_t)B_ * H_ * 64 * 64 * 4);
  float*  kvb     = (float*)carve((size_t)B_ * H_ * 64 * 128 * 4);
  __bf16* w2tbf   = (__bf16*)carve((size_t)B_ * H_ * 64 * 128 * 2);

  // 1) bf16 casts (x plain; weights cast + transposed to [N][K])
  cast_f32_bf16<<<4096, 256, 0, stream>>>(x, xbf, NT * 2048);
  dim3 tb(32, 8);
  cast_transpose_bf16<<<dim3(2048 / 32, 2048 / 32), tb, 0, stream>>>(Wq, wqt, 2048, 2048);
  cast_transpose_bf16<<<dim3(2048 / 32, 1024 / 32), tb, 0, stream>>>(Wk, wkt, 2048, 1024);
  cast_transpose_bf16<<<dim3(2048 / 32, 1024 / 32), tb, 0, stream>>>(Wv, wvt, 2048, 1024);
  cast_transpose_bf16<<<dim3(2048 / 32, 2048 / 32), tb, 0, stream>>>(Wo, wot, 2048, 2048);

  // 2) q/k/v projections (WMMA bf16 GEMMs, async double-buffered pipeline)
  gemm_bf16_f32<<<dim3(2048 / 64, (int)(NT / 64)), 128, 0, stream>>>(xbf, wqt, qf, (int)NT, 2048, 2048);
  gemm_bf16_f32<<<dim3(1024 / 64, (int)(NT / 64)), 128, 0, stream>>>(xbf, wkt, kf, (int)NT, 1024, 2048);
  gemm_bf16_f32<<<dim3(1024 / 64, (int)(NT / 64)), 128, 0, stream>>>(xbf, wvt, vf, (int)NT, 1024, 2048);

  // 3) landmarks + kernel_proj
  landmark_proj<<<dim3(H_,   B_), 128, 0, stream>>>(qf, H_,   Wkp1, Wkp2, qland, (__bf16*)nullptr);
  landmark_proj<<<dim3(HKV_, B_), 128, 0, stream>>>(kf, HKV_, Wkp1, Wkp2, kland, klandbf);

  // 4) K_mm softmax + Newton pinv (6 iters)
  kmm_pinv<<<dim3(H_, B_), 256, 0, stream>>>(qland, kland, zinv);

  // 5) KV = softmax(K_mk)·v, flash-streamed over T
  kmk_softmax_v<<<dim3(64, H_, B_), 128, 0, stream>>>(kf, vf, qland, kvb);

  // 6) W2ᵀ = (Zinv·KV)ᵀ (bf16, pre-transposed for the fused kernel)
  apply_pinv<<<dim3(H_, B_), 256, 0, stream>>>(zinv, kvb, w2tbf);

  // 7) fused attn = softmax(Q·K_landᵀ)·W2 (WMMA x2), bf16 out
  kqm_out<<<dim3(T_ / 64, H_, B_), 128, 0, stream>>>(qf, klandbf, w2tbf, attnbf);

  // 8) output projection (WMMA bf16 GEMM) -> d_out f32
  gemm_bf16_f32<<<dim3(2048 / 64, (int)(NT / 64)), 128, 0, stream>>>(attnbf, wot, out, (int)NT, 2048, 2048);
}
